// RNNQNetworkZeroState_7885559955596
// MI455X (gfx1250) — matrix operand order repaired and checked by reading, not compiled, not run on hardware
//
#include <hip/hip_runtime.h>

typedef __attribute__((ext_vector_type(16))) _Float16 v16h;
typedef __attribute__((ext_vector_type(8)))  _Float16 v8h;
typedef __attribute__((ext_vector_type(8)))  float    v8f;

typedef int i32x4 __attribute__((vector_size(16)));   // native int4 for async builtins

#define BB     16
#define SS     2048
#define D_IN   1024
#define D_H    2048
#define D_ACT  1024
#define M_TOT  (BB * SS)        // 32768 rows

#define TM 128
#define TN 128
#define TK 32
#define THREADS 256

#if __has_builtin(__builtin_amdgcn_global_load_async_to_lds_b128) && \
    __has_builtin(__builtin_amdgcn_global_store_async_from_lds_b128) && \
    __has_builtin(__builtin_amdgcn_s_wait_asynccnt)
#define HAS_ASYNC_LDS 1
#else
#define HAS_ASYNC_LDS 0
#endif

#if HAS_ASYNC_LDS
__device__ __forceinline__ void async_ld16(const _Float16* g, _Float16* l) {
    __builtin_amdgcn_global_load_async_to_lds_b128((i32x4*)g, (i32x4*)l, 0, 0);
}
__device__ __forceinline__ void async_st16(_Float16* g, const _Float16* l) {
    __builtin_amdgcn_global_store_async_from_lds_b128((i32x4*)g, (i32x4*)l, 0, 0);
}
#endif

// ---------------- f32 -> f16 conversion (bandwidth-trivial pre-pass) ----------------
__global__ __launch_bounds__(256) void cvt_f32_to_f16(const float* __restrict__ src,
                                                      _Float16* __restrict__ dst,
                                                      long long n) {
    long long i = ((long long)blockIdx.x * blockDim.x + threadIdx.x) * 4;
    if (i + 3 < n) {
        float4 v = *(const float4*)(src + i);
        dst[i + 0] = (_Float16)v.x;
        dst[i + 1] = (_Float16)v.y;
        dst[i + 2] = (_Float16)v.z;
        dst[i + 3] = (_Float16)v.w;
    }
}

// ---------------- tiled WMMA GEMM: C[M,N] = act(A[M,K] * W[N,K]^T + bias (+bias2)) ----
// A: f16 row-major [M][K]; W: f16 row-major [N][K] (== B column-major, WMMA-friendly)
// Async global->LDS triple-buffered staging (branch-free steady loop, peeled tail);
// async LDS->global coalesced epilogue.
template <int RELU, int OUT_F16, int HAS_B2>
__global__ __launch_bounds__(256) void gemm_wmma(
    const _Float16* __restrict__ A,
    const _Float16* __restrict__ W,
    const float*    __restrict__ bias,
    const float*    __restrict__ bias2,
    _Float16*       __restrict__ Cf16,
    float*          __restrict__ Cf32,
    int M, int N, int K)
{
    constexpr int STAGE_HALF  = TM * TK + TN * TK;            // halves per buffer (8192)
#if HAS_ASYNC_LDS
    constexpr int NBUF = 3;
#else
    constexpr int NBUF = 2;
#endif
    constexpr int STAGE_BYTES = NBUF * STAGE_HALF * 2;
    constexpr int OUT_BYTES   = OUT_F16 ? TM * TN * 2 : TM * TN * 4;
    constexpr int SMEM_BYTES  = STAGE_BYTES > OUT_BYTES ? STAGE_BYTES : OUT_BYTES;
    __shared__ alignas(32) char smem[SMEM_BYTES];

    const int tid      = threadIdx.x;
    const int lane     = tid & 31;
    const int wave     = tid >> 5;
    const int wm       = wave & 3;        // 4 M-slices of 32 rows
    const int wn       = wave >> 2;       // 2 N-slices of 64 cols
    const int lane16   = lane & 15;
    const int laneHalf = lane >> 4;       // 0 or 1

    const int m0 = blockIdx.y * TM;
    const int n0 = blockIdx.x * TN;

    // Staging assignment: chunk c in [0,512): row = c>>2, col(halves) = (c&3)*8
    const int rA0 = tid >> 2,           cA0 = (tid & 3) * 8;
    const int rA1 = (tid + 256) >> 2,   cA1 = ((tid + 256) & 3) * 8;
    const _Float16* gA0 = A + (size_t)(m0 + rA0) * K + cA0;
    const _Float16* gA1 = A + (size_t)(m0 + rA1) * K + cA1;
    const _Float16* gW0 = W + (size_t)(n0 + rA0) * K + cA0;
    const _Float16* gW1 = W + (size_t)(n0 + rA1) * K + cA1;
    const int sA0 = rA0 * TK + cA0;
    const int sA1 = rA1 * TK + cA1;
    const int sW0 = TM * TK + rA0 * TK + cA0;
    const int sW1 = TM * TK + rA1 * TK + cA1;

    v8f acc[2][4];
#pragma unroll
    for (int i = 0; i < 2; ++i)
#pragma unroll
        for (int j = 0; j < 4; ++j)
            acc[i][j] = (v8f){0.f, 0.f, 0.f, 0.f, 0.f, 0.f, 0.f, 0.f};

    // Shared 8-WMMA tile compute (always inlined at both call sites)
    auto compute_tile = [&](const _Float16* sbA) __attribute__((always_inline)) {
        const _Float16* sbB = sbA + TM * TK;
        v16h afrag[2];
#pragma unroll
        for (int i = 0; i < 2; ++i) {
            const _Float16* base = sbA + (wm * 32 + i * 16 + lane16) * TK;
            v8h lo = *(const v8h*)(base + laneHalf * 16);
            v8h hi = *(const v8h*)(base + laneHalf * 16 + 8);
            afrag[i] = __builtin_shufflevector(lo, hi,
                0, 1, 2, 3, 4, 5, 6, 7, 8, 9, 10, 11, 12, 13, 14, 15);
        }
        v16h bfrag[4];
#pragma unroll
        for (int j = 0; j < 4; ++j) {
            const _Float16* base = sbB + (wn * 64 + j * 16 + lane16) * TK + laneHalf * 16;
            bfrag[j] = *(const v16h*)base;
        }
#pragma unroll
        for (int i = 0; i < 2; ++i)
#pragma unroll
            for (int j = 0; j < 4; ++j)
                acc[i][j] = __builtin_amdgcn_wmma_f32_16x16x32_f16(
                    false, afrag[i], false, bfrag[j],
                    (short)0, acc[i][j], false, false);
    };

#if HAS_ASYNC_LDS
    // ---- async triple-buffered pipeline: branch-free steady loop, peeled tail ----
    {
        _Float16* sb = (_Float16*)smem;   // buffer 0 <- tile k=0
        async_ld16(gA0, sb + sA0);
        async_ld16(gA1, sb + sA1);
        async_ld16(gW0, sb + sW0);
        async_ld16(gW1, sb + sW1);
    }
    const _Float16* pA0 = gA0 + TK;   // next-tile pointers, bumped by TK per iter
    const _Float16* pA1 = gA1 + TK;
    const _Float16* pW0 = gW0 + TK;
    const _Float16* pW1 = gW1 + TK;
    int cur = 0;
    for (int k0 = 0; k0 < K - TK; k0 += TK) {
        const int nxt = (cur == NBUF - 1) ? 0 : cur + 1;
        _Float16* nb = (_Float16*)smem + nxt * STAGE_HALF;  // read 2 barriers ago
        async_ld16(pA0, nb + sA0);
        async_ld16(pA1, nb + sA1);
        async_ld16(pW0, nb + sW0);
        async_ld16(pW1, nb + sW1);
        pA0 += TK; pA1 += TK; pW0 += TK; pW1 += TK;
        __builtin_amdgcn_s_wait_asynccnt(4);   // current tile complete
        __syncthreads();
        compute_tile((const _Float16*)smem + cur * STAGE_HALF);
        cur = nxt;
    }
    // tail: last tile, nothing left to prefetch
    __builtin_amdgcn_s_wait_asynccnt(0);
    __syncthreads();
    compute_tile((const _Float16*)smem + cur * STAGE_HALF);
#else
    // ---- fallback: register-prefetch double buffer ----
    uint4 pa0 = *(const uint4*)gA0;
    uint4 pa1 = *(const uint4*)gA1;
    uint4 pw0 = *(const uint4*)gW0;
    uint4 pw1 = *(const uint4*)gW1;
    {
        _Float16* sb = (_Float16*)smem;
        *(uint4*)(sb + sA0) = pa0;  *(uint4*)(sb + sA1) = pa1;
        *(uint4*)(sb + sW0) = pw0;  *(uint4*)(sb + sW1) = pw1;
    }
    for (int k0 = 0; k0 < K; k0 += TK) {
        const int cur = (k0 / TK) & 1;
        __syncthreads();
        const bool more = (k0 + TK) < K;
        if (more) {
            pa0 = *(const uint4*)(gA0 + k0 + TK);
            pa1 = *(const uint4*)(gA1 + k0 + TK);
            pw0 = *(const uint4*)(gW0 + k0 + TK);
            pw1 = *(const uint4*)(gW1 + k0 + TK);
        }
        compute_tile((const _Float16*)smem + cur * STAGE_HALF);
        if (more) {
            _Float16* nb = (_Float16*)smem + (cur ^ 1) * STAGE_HALF;
            *(uint4*)(nb + sA0) = pa0;  *(uint4*)(nb + sA1) = pa1;
            *(uint4*)(nb + sW0) = pw0;  *(uint4*)(nb + sW1) = pw1;
        }
    }
#endif

    // ---------- Epilogue: bias + activation -> LDS tile -> coalesced stores ----------
    float bsum[4];
#pragma unroll
    for (int j = 0; j < 4; ++j) {
        const int ncol = n0 + wn * 64 + j * 16 + lane16;
        bsum[j] = bias[ncol] + (HAS_B2 ? bias2[ncol] : 0.f);
    }

    __syncthreads();   // all frag reads done; safe to overwrite staging LDS

    if (OUT_F16) {
        _Float16* ot = (_Float16*)smem;
#pragma unroll
        for (int i = 0; i < 2; ++i) {
            const int rbase = wm * 32 + i * 16 + laneHalf * 8;
#pragma unroll
            for (int j = 0; j < 4; ++j) {
                const int col = wn * 64 + j * 16 + lane16;
#pragma unroll
                for (int r = 0; r < 8; ++r) {
                    float v = acc[i][j][r] + bsum[j];
                    if (RELU) v = v > 0.f ? v : 0.f;
                    ot[(rbase + r) * TN + col] = (_Float16)v;
                }
            }
        }
        __syncthreads();
#pragma unroll
        for (int p = 0; p < 8; ++p) {   // 32KB tile, 16B chunks, fully coalesced
            const int chunk = tid + p * THREADS;
            const int row = chunk >> 4;
            const int col = (chunk & 15) * 8;
            _Float16* dst = Cf16 + (size_t)(m0 + row) * N + n0 + col;
#if HAS_ASYNC_LDS
            async_st16(dst, ot + row * TN + col);
#else
            *(uint4*)dst = *(const uint4*)(ot + row * TN + col);
#endif
        }
    } else {
        float* ot = (float*)smem;
#pragma unroll
        for (int i = 0; i < 2; ++i) {
            const int rbase = wm * 32 + i * 16 + laneHalf * 8;
#pragma unroll
            for (int j = 0; j < 4; ++j) {
                const int col = wn * 64 + j * 16 + lane16;
#pragma unroll
                for (int r = 0; r < 8; ++r) {
                    float v = acc[i][j][r] + bsum[j];
                    if (RELU) v = v > 0.f ? v : 0.f;
                    ot[(rbase + r) * TN + col] = v;
                }
            }
        }
        __syncthreads();
#pragma unroll
        for (int p = 0; p < 16; ++p) {  // 64KB tile, 16B chunks, fully coalesced
            const int chunk = tid + p * THREADS;
            const int row = chunk >> 5;
            const int col = (chunk & 31) * 4;
            float* dst = Cf32 + (size_t)(m0 + row) * N + n0 + col;
#if HAS_ASYNC_LDS
            async_st16((_Float16*)dst, (const _Float16*)(ot + row * TN + col));
#else
            *(uint4*)dst = *(const uint4*)(ot + row * TN + col);
#endif
        }
    }
    // async stores drained by S_ENDPGM's implicit wait-idle
}

extern "C" void kernel_launch(void* const* d_in, const int* in_sizes, int n_in,
                              void* d_out, int out_size, void* d_ws, size_t ws_size,
                              hipStream_t stream) {
    (void)in_sizes; (void)n_in; (void)out_size; (void)ws_size;

    const float* x       = (const float*)d_in[0];
    const float* W_init  = (const float*)d_in[1];
    const float* b_init  = (const float*)d_in[2];
    const float* W_ih    = (const float*)d_in[3];
    const float* b_ih    = (const float*)d_in[4];
    const float* b_hh    = (const float*)d_in[5];
    const float* W_final = (const float*)d_in[6];
    const float* b_final = (const float*)d_in[7];
    float* out = (float*)d_out;

    // Workspace layout (f16 staging)
    char* ws = (char*)d_ws;
    size_t off = 0;
    _Float16* x_h  = (_Float16*)(ws + off); off += (size_t)M_TOT * D_IN  * 2;
    _Float16* Wi_h = (_Float16*)(ws + off); off += (size_t)D_H   * D_IN  * 2;
    _Float16* Wh_h = (_Float16*)(ws + off); off += (size_t)D_H   * D_H   * 2;
    _Float16* Wf_h = (_Float16*)(ws + off); off += (size_t)D_ACT * D_H   * 2;
    _Float16* h_h  = (_Float16*)(ws + off); off += (size_t)M_TOT * D_H   * 2;
    _Float16* o_h  = (_Float16*)(ws + off); off += (size_t)M_TOT * D_H   * 2;

    auto launch_cvt = [&](const float* s, _Float16* d, long long n) {
        long long blocks = n / (4LL * 256);
        cvt_f32_to_f16<<<dim3((unsigned)blocks), dim3(256), 0, stream>>>(s, d, n);
    };
    launch_cvt(x,       x_h,  (long long)M_TOT * D_IN);
    launch_cvt(W_init,  Wi_h, (long long)D_H   * D_IN);
    launch_cvt(W_ih,    Wh_h, (long long)D_H   * D_H);
    launch_cvt(W_final, Wf_h, (long long)D_ACT * D_H);

    dim3 blk(THREADS);
    // GEMM1: h = relu(x * W_init^T + b_init)
    gemm_wmma<1, 1, 0><<<dim3(D_H / TN, M_TOT / TM), blk, 0, stream>>>(
        x_h, Wi_h, b_init, nullptr, h_h, nullptr, M_TOT, D_H, D_IN);
    // GEMM2: o = relu(h * W_ih^T + b_ih + b_hh)
    gemm_wmma<1, 1, 1><<<dim3(D_H / TN, M_TOT / TM), blk, 0, stream>>>(
        h_h, Wh_h, b_ih, b_hh, o_h, nullptr, M_TOT, D_H, D_H);
    // GEMM3: y = o * W_final^T + b_final -> f32 out
    gemm_wmma<0, 0, 0><<<dim3(D_ACT / TN, M_TOT / TM), blk, 0, stream>>>(
        o_h, Wf_h, b_final, nullptr, nullptr, out, M_TOT, D_ACT, D_H);
}
